// Attention_20023137534494
// MI455X (gfx1250) — matrix operand order repaired and checked by reading, not compiled, not run on hardware
//
#include <hip/hip_runtime.h>
#include <hip/hip_bf16.h>

// ---------------- problem constants ----------------
constexpr int B_ = 2;
constexpr int S_ = 2048;
constexpr int D_ = 2048;
constexpr int NH = 16;
constexpr int HD = 128;          // D_/NH

typedef __attribute__((ext_vector_type(16))) __bf16 v16bf;
typedef __attribute__((ext_vector_type(8)))  __bf16 v8bf;
typedef __attribute__((ext_vector_type(8)))  float  v8f;
typedef __attribute__((ext_vector_type(4)))  float  v4f;

union V16U { v16bf v; v8bf h[2]; };

// f32 -> bf16, round-to-nearest-even (bit manip; independent of cast lowering)
__device__ __forceinline__ __bf16 f2bf(float f) {
  unsigned u = __builtin_bit_cast(unsigned, f);
  u += 0x7FFFu + ((u >> 16) & 1u);
  unsigned short h = (unsigned short)(u >> 16);
  return __builtin_bit_cast(__bf16, h);
}

__device__ __forceinline__ v8f wmma_bf16(v16bf a, v16bf b, v8f c) {
  // D = A(16x32) * B(32x16) + C, f32 accumulate
  return __builtin_amdgcn_wmma_f32_16x16x32_bf16(
      /*neg_a=*/false, a, /*neg_b=*/false, b,
      /*c_mod=*/(short)0, c, /*reuse_a=*/false, /*reuse_b=*/false);
}

// A-fragment (16x32 bf16, row-major source, ld elements per row):
//   lanes 0-15 : M=lane,   K = 0..15 contiguous
//   lanes16-31 : M=lane-16,K = 16..31 contiguous
__device__ __forceinline__ v16bf load_a_frag(const __bf16* base, int ld, int lane) {
  int r    = lane & 15;
  int koff = (lane >> 4) << 4;         // 0 or 16
  const __bf16* p = base + (size_t)r * ld + koff;
  V16U u;
  u.h[0] = *(const v8bf*)(p);
  u.h[1] = *(const v8bf*)(p + 8);
  return u.v;
}

// B-fragment (32x16 bf16). Source stored "NT": row = N column, K contiguous.
//   lanes 0-15 : N=lane,   K = 0..15 contiguous
//   lanes16-31 : N=lane-16,K = 16..31 contiguous
__device__ __forceinline__ v16bf load_b_frag(const __bf16* base, int ld, int lane) {
  int n    = lane & 15;
  int koff = (lane >> 4) << 4;         // 0 or 16
  return *(const v16bf*)(base + (size_t)n * ld + koff);
}

// ---------------- kernel 1: f32 -> bf16 convert, 8 elems/thread ----------------
__global__ void cvt_f32_bf16(const float* __restrict__ in, __bf16* __restrict__ out, int n) {
  int i = (blockIdx.x * blockDim.x + threadIdx.x) * 8;
  if (i + 8 > n) return;
  v4f f0 = *(const v4f*)(in + i);
  v4f f1 = *(const v4f*)(in + i + 4);
  v8bf o;
#pragma unroll
  for (int j = 0; j < 4; ++j) { o[j] = f2bf(f0[j]); o[4 + j] = f2bf(f1[j]); }
  *(v8bf*)(out + i) = o;
}

// ---------------- kernel 2: NT bf16 GEMM, f32 out ----------------
// C[M,N] = A[M,K] * B[N,K]^T   (both row-major, K contiguous)
// block = 256 threads = 8 waves (4x2), wave tile 32x64, block tile 128x128
__global__ __launch_bounds__(256) void gemm_nt_bf16(
    const __bf16* __restrict__ A, const __bf16* __restrict__ Bm,
    float* __restrict__ C, int M, int N, int K) {
  int lane = threadIdx.x & 31;
  int w    = threadIdx.x >> 5;
  int wm   = w >> 1;                   // 0..3
  int wn   = w & 1;                    // 0..1
  int m0   = blockIdx.y * 128 + wm * 32;
  int n0   = blockIdx.x * 128 + wn * 64;

  v8f acc[2][4] = {};

  for (int k0 = 0; k0 < K; k0 += 32) {
    v16bf a0 = load_a_frag(A + (size_t)m0 * K + k0, K, lane);
    v16bf a1 = load_a_frag(A + (size_t)(m0 + 16) * K + k0, K, lane);
    v16bf b0 = load_b_frag(Bm + (size_t)(n0 +  0) * K + k0, K, lane);
    v16bf b1 = load_b_frag(Bm + (size_t)(n0 + 16) * K + k0, K, lane);
    v16bf b2 = load_b_frag(Bm + (size_t)(n0 + 32) * K + k0, K, lane);
    v16bf b3 = load_b_frag(Bm + (size_t)(n0 + 48) * K + k0, K, lane);
    acc[0][0] = wmma_bf16(a0, b0, acc[0][0]);
    acc[0][1] = wmma_bf16(a0, b1, acc[0][1]);
    acc[0][2] = wmma_bf16(a0, b2, acc[0][2]);
    acc[0][3] = wmma_bf16(a0, b3, acc[0][3]);
    acc[1][0] = wmma_bf16(a1, b0, acc[1][0]);
    acc[1][1] = wmma_bf16(a1, b1, acc[1][1]);
    acc[1][2] = wmma_bf16(a1, b2, acc[1][2]);
    acc[1][3] = wmma_bf16(a1, b3, acc[1][3]);
  }

  // C/D layout: VGPR e -> row m + half*8 + e, lane -> col n
  int half = lane >> 4, nn = lane & 15;
#pragma unroll
  for (int i = 0; i < 2; ++i)
#pragma unroll
    for (int j = 0; j < 4; ++j)
#pragma unroll
      for (int e = 0; e < 8; ++e) {
        size_t row = (size_t)(m0 + i * 16 + half * 8 + e);
        C[row * N + n0 + j * 16 + nn] = acc[i][j][e];
      }
}

// ---------------- kernel 3: rotary + pack to [B,H,S,HD] bf16 ----------------
__global__ void rotary_pack(const float* __restrict__ qkv,
                            const float* __restrict__ cosb,
                            const float* __restrict__ sinb,
                            __bf16* __restrict__ Q, __bf16* __restrict__ K,
                            __bf16* __restrict__ V) {
  int idx = blockIdx.x * blockDim.x + threadIdx.x;    // B*S*NH*(HD/2) = 2^22
  int i = idx & 63;                                   // pair index in head
  int h = (idx >> 6) & 15;
  int s = (idx >> 10) & 2047;
  int b = idx >> 21;

  const float* row = qkv + (size_t)(b * S_ + s) * (3 * D_);
  int col = h * HD + 2 * i;
  float qr = row[col],          qi = row[col + 1];
  float kr = row[D_ + col],     ki = row[D_ + col + 1];
  float vr = row[2 * D_ + col], vi = row[2 * D_ + col + 1];
  float c  = cosb[s * 64 + i];
  float sn = sinb[s * 64 + i];

  size_t o = ((size_t)(b * NH + h) * S_ + s) * HD + 2 * i;
  Q[o]     = f2bf(qr * c - qi * sn);
  Q[o + 1] = f2bf(qr * sn + qi * c);
  K[o]     = f2bf(kr * c - ki * sn);
  K[o + 1] = f2bf(kr * sn + ki * c);
  V[o]     = f2bf(vr);
  V[o + 1] = f2bf(vi);
}

// ---------------- kernel 4: flash attention ----------------
// grid (S/64, B*NH), block 128 (4 waves). Wave w owns queries [q0+16w, q0+16w+16).
__global__ __launch_bounds__(128) void flash_attn(
    const __bf16* __restrict__ Q, const __bf16* __restrict__ K,
    const __bf16* __restrict__ V, __bf16* __restrict__ O) {
  __shared__ __bf16 Ksh[32 * 128];      // K tile, row-major [k][d]
  __shared__ __bf16 Vtsh[128 * 32];     // V tile transposed [d][k]
  __shared__ __bf16 Psh[4 * 16 * 32];   // per-wave P staging

  int lane = threadIdx.x & 31;
  int w    = threadIdx.x >> 5;
  int bh   = blockIdx.y;                // b*NH + h
  int q0   = blockIdx.x * 64;
  int qt   = q0 + w * 16;

  const __bf16* Qb = Q + ((size_t)bh * S_ + qt) * HD;
  const __bf16* Kb = K + (size_t)bh * S_ * HD;
  const __bf16* Vb = V + (size_t)bh * S_ * HD;

  v16bf qa[4];
#pragma unroll
  for (int t = 0; t < 4; ++t) qa[t] = load_a_frag(Qb + 32 * t, HD, lane);

  v8f o[8] = {};
  float m[8], l[8];
#pragma unroll
  for (int e = 0; e < 8; ++e) { m[e] = -1e30f; l[e] = 0.0f; }

  int half = lane >> 4, nn = lane & 15;
  const float scale = 0.0883883476f;    // 1/sqrt(128)
  __bf16* Pw = Psh + w * (16 * 32);

  for (int ks = 0; ks < q0 + 64; ks += 32) {
    // cooperative staging of 32 keys x 128 dims, 128-bit granularity
    // 512 v8bf chunks over 128 threads -> 4 per thread
#pragma unroll
    for (int t = 0; t < 4; ++t) {
      int c  = threadIdx.x + t * 128;   // 0..511
      int kk = c >> 4;                  // key row 0..31
      int dc = (c & 15) << 3;           // d chunk 0,8,...,120
      v8bf kv = *(const v8bf*)(Kb + (size_t)(ks + kk) * HD + dc);
      v8bf vv = *(const v8bf*)(Vb + (size_t)(ks + kk) * HD + dc);
      *(v8bf*)(Ksh + kk * 128 + dc) = kv;             // ds_store_b128
#pragma unroll
      for (int j = 0; j < 8; ++j)                     // register-side transpose
        Vtsh[(dc + j) * 32 + kk] = vv[j];
    }
    __syncthreads();

    if (ks <= qt + 15) {                // wave-uniform: any unmasked key in tile
      // scores S = Q * K^T : 16x32 in two 16x16 accumulators
      v8f s0 = {}, s1 = {};
#pragma unroll
      for (int t = 0; t < 4; ++t) {
        v16bf kb0 = load_b_frag(Ksh + 32 * t,            128, lane);
        v16bf kb1 = load_b_frag(Ksh + 16 * 128 + 32 * t, 128, lane);
        s0 = wmma_bf16(qa[t], kb0, s0);
        s1 = wmma_bf16(qa[t], kb1, s1);
      }

      // online softmax in accumulator layout (VGPR e = row, lane = key)
      float alpha[8];
#pragma unroll
      for (int e = 0; e < 8; ++e) {
        int qrow = qt + half * 8 + e;
        float v0 = s0[e] * scale;
        float v1 = s1[e] * scale;
        if (ks + nn > qrow)      v0 = -1e30f;   // causal mask
        if (ks + 16 + nn > qrow) v1 = -1e30f;
        float t = fmaxf(v0, v1);
        t = fmaxf(t, __shfl_xor(t, 1, 32));
        t = fmaxf(t, __shfl_xor(t, 2, 32));
        t = fmaxf(t, __shfl_xor(t, 4, 32));
        t = fmaxf(t, __shfl_xor(t, 8, 32));
        float mn = fmaxf(m[e], t);
        float a  = __expf(m[e] - mn);
        v0 = __expf(v0 - mn);
        v1 = __expf(v1 - mn);
        float rs = v0 + v1;
        rs += __shfl_xor(rs, 1, 32);
        rs += __shfl_xor(rs, 2, 32);
        rs += __shfl_xor(rs, 4, 32);
        rs += __shfl_xor(rs, 8, 32);
        l[e] = l[e] * a + rs;
        m[e] = mn;
        alpha[e] = a;
        s0[e] = v0;
        s1[e] = v1;
      }
#pragma unroll
      for (int j = 0; j < 8; ++j) {
        v8f t = o[j];
#pragma unroll
        for (int e = 0; e < 8; ++e) t[e] *= alpha[e];
        o[j] = t;
      }

      // re-layout P (accumulator layout) -> A-fragment via per-wave LDS tile
#pragma unroll
      for (int e = 0; e < 8; ++e) {
        int r = half * 8 + e;
        Pw[r * 32 + nn]      = f2bf(s0[e]);
        Pw[r * 32 + 16 + nn] = f2bf(s1[e]);
      }
      v16bf pa = load_a_frag(Pw, 32, lane);   // same-wave DS ordering

      // O += P(16x32) * V(32x128), 8 d-tiles of 16
#pragma unroll
      for (int j = 0; j < 8; ++j) {
        v16bf vb = load_b_frag(Vtsh + (j * 16) * 32, 32, lane);
        o[j] = wmma_bf16(pa, vb, o[j]);
      }
    }
    __syncthreads();
  }

  // normalize and write bf16 output in [B,S,D] layout
  int b = bh >> 4, h = bh & 15;
#pragma unroll
  for (int j = 0; j < 8; ++j)
#pragma unroll
    for (int e = 0; e < 8; ++e) {
      int qrow = qt + half * 8 + e;
      size_t oi = ((size_t)(b * S_ + qrow)) * D_ + h * HD + j * 16 + nn;
      O[oi] = f2bf(o[j][e] / l[e]);
    }
}

// ---------------- host-side launcher ----------------
extern "C" void kernel_launch(void* const* d_in, const int* in_sizes, int n_in,
                              void* d_out, int out_size, void* d_ws, size_t ws_size,
                              hipStream_t stream) {
  (void)in_sizes; (void)n_in; (void)out_size; (void)ws_size;
  const float* x     = (const float*)d_in[0];
  const float* fcos  = (const float*)d_in[1];
  const float* fsin  = (const float*)d_in[2];
  // d_in[3] = mask (causality handled analytically), d_in[6] = input_pos (==0)
  const float* wqkv  = (const float*)d_in[4];
  const float* wo    = (const float*)d_in[5];
  float* out = (float*)d_out;

  const size_t nX    = (size_t)B_ * S_ * D_;        // 8,388,608
  const size_t nWqkv = (size_t)3 * D_ * D_;         // 12,582,912
  const size_t nWo   = (size_t)D_ * D_;             // 4,194,304
  const size_t nQKV  = (size_t)B_ * S_ * 3 * D_;    // 25,165,824

  char* ws = (char*)d_ws;
  size_t off = 0;
  auto take = [&](size_t bytes) { char* p = ws + off; off = (off + bytes + 255) & ~(size_t)255; return p; };

  float*  qkv_f  = (float*)take(nQKV * 4);          // 100.7 MB
  __bf16* xb     = (__bf16*)take(nX * 2);
  __bf16* wqkvb  = (__bf16*)take(nWqkv * 2);
  __bf16* wob    = (__bf16*)take(nWo * 2);
  __bf16* qb     = (__bf16*)take(nX * 2);           // [B,H,S,HD]
  __bf16* kb     = (__bf16*)take(nX * 2);
  __bf16* vb     = (__bf16*)take(nX * 2);
  __bf16* aob    = (__bf16*)take(nX * 2);           // [B,S,D]

  // 1) converts (8 elems/thread, b128 in/out)
  cvt_f32_bf16<<<(int)(nX    / (256 * 8)), 256, 0, stream>>>(x,    xb,    (int)nX);
  cvt_f32_bf16<<<(int)(nWqkv / (256 * 8)), 256, 0, stream>>>(wqkv, wqkvb, (int)nWqkv);
  cvt_f32_bf16<<<(int)(nWo   / (256 * 8)), 256, 0, stream>>>(wo,   wob,   (int)nWo);

  // 2) QKV projection: [4096,6144] = x[4096,2048] * wqkv[6144,2048]^T
  gemm_nt_bf16<<<dim3((3 * D_) / 128, (B_ * S_) / 128), 256, 0, stream>>>(
      xb, wqkvb, qkv_f, B_ * S_, 3 * D_, D_);

  // 3) rotary + pack per-head bf16
  rotary_pack<<<(B_ * S_ * NH * (HD / 2)) / 256, 256, 0, stream>>>(
      qkv_f, fcos, fsin, qb, kb, vb);

  // 4) causal flash attention
  flash_attn<<<dim3(S_ / 64, B_ * NH), 128, 0, stream>>>(qb, kb, vb, aob);

  // 5) output projection: [4096,2048] = attn[4096,2048] * wo[2048,2048]^T
  gemm_nt_bf16<<<dim3(D_ / 128, (B_ * S_) / 128), 256, 0, stream>>>(
      aob, wob, out, B_ * S_, D_, D_);
}
